// VanillaAttention_65068754535033
// MI455X (gfx1250) — matrix operand order repaired and checked by reading, not compiled, not run on hardware
//
#include <hip/hip_runtime.h>
#include <hip/hip_bf16.h>
#include <math.h>

#define B_   2
#define T_   2048
#define D_   2048
#define NH   16
#define NKV  4
#define HD   128

typedef __attribute__((ext_vector_type(16))) __bf16 v16bf;
typedef __attribute__((ext_vector_type(8)))  float  v8f;

union FragAB { v16bf v; unsigned u[8]; };

__device__ inline unsigned short f2bf(float f) {
  union { float f; unsigned u; } x; x.f = f;
  unsigned r = x.u + 0x7fffu + ((x.u >> 16) & 1u);   // round-to-nearest-even
  return (unsigned short)(r >> 16);
}
__device__ inline float bf2f(unsigned short h) {
  union { unsigned u; float f; } x; x.u = ((unsigned)h) << 16;
  return x.f;
}
__device__ inline v8f vzero8() {
  v8f z = {0.f,0.f,0.f,0.f,0.f,0.f,0.f,0.f};
  return z;
}
__device__ inline v8f wmma_bf16(const FragAB& a, const FragAB& b, v8f c) {
  return __builtin_amdgcn_wmma_f32_16x16x32_bf16(
      /*neg_a=*/false, a.v, /*neg_b=*/false, b.v,
      /*c_mod=*/(short)0, c, /*reuse_a=*/false, /*reuse_b=*/false);
}

// Low 32 bits of a generic pointer to __shared__ == addrspace(3) byte offset
// used by the DS / async-LDS units.
__device__ inline unsigned lds_off(const void* p) {
  return (unsigned)(unsigned long long)p;
}
// CDNA5 async copy: 16B per lane, global (SGPR base + per-lane voffset) -> LDS.
// Tracked by ASYNCcnt.
__device__ inline void async_b128(unsigned lds_addr, const void* sbase, unsigned voff) {
  asm volatile("global_load_async_to_lds_b128 %0, %1, %2"
               :: "v"(lds_addr), "v"(voff), "s"(sbase) : "memory");
}
__device__ inline void wait_async0() {
  asm volatile("s_wait_asynccnt 0" ::: "memory");
}

// ---------------------------------------------------------------- fp32 -> bf16
__global__ __launch_bounds__(256) void k_f32_to_bf16(const float* __restrict__ in,
                                                     unsigned short* __restrict__ out,
                                                     int n) {
  int i = (blockIdx.x * 256 + threadIdx.x) * 4;
  if (i < n) {
    float4 v = *(const float4*)(in + i);
    out[i + 0] = f2bf(v.x); out[i + 1] = f2bf(v.y);
    out[i + 2] = f2bf(v.z); out[i + 3] = f2bf(v.w);
  }
}

// fp32 [K][N] -> bf16 transposed [N][K] (weights: done once, reused by all tiles)
__global__ __launch_bounds__(256) void k_f32_to_bf16_T(const float* __restrict__ in,
                                                       unsigned short* __restrict__ out,
                                                       int N, int K) {
  int idx = blockIdx.x * 256 + threadIdx.x;
  if (idx < N * K) {
    int k = idx / N, n = idx - k * N;       // coalesced read along N
    out[(size_t)n * K + k] = f2bf(in[idx]);
  }
}

// ---------------------------------------------------------------- bf16 GEMM
// C(MxN) = A(MxK, row-major) * Wt(NxK, row-major i.e. W transposed), fp32 acc.
// Block tile 128x128, 8 waves, wave tile 32x64. Double-buffered LDS filled with
// global_load_async_to_lds_b128; staging of tile t+1 overlaps WMMA of tile t.
// MODE 0: Q [b][h][t][hd] | 1: K [b][hkv][t][hd] | 2: V^T [b][hkv][hd][t] | 3: fp32 [M][N]
template<int MODE>
__global__ __launch_bounds__(256) void k_gemm(const unsigned short* __restrict__ A,
                                              const unsigned short* __restrict__ Wt,
                                              void* __restrict__ outp,
                                              int N, int K) {
  __shared__ unsigned short As[2][128 * 40];   // [m][k], stride 40 (80B: 16B-aligned rows)
  __shared__ unsigned short Bs[2][128 * 40];   // [n][k], stride 40

  const int tid  = threadIdx.x;
  const int lane = tid & 31, w = tid >> 5;
  const int half = lane >> 4, lm = lane & 15;
  const int wm = w & 3, wn = w >> 2;
  const int n0 = blockIdx.x * 128, m0 = blockIdx.y * 128;

  v8f acc[2][4];
#pragma unroll
  for (int i = 0; i < 2; ++i)
#pragma unroll
    for (int j = 0; j < 4; ++j) acc[i][j] = vzero8();

  int koff[8];
#pragma unroll
  for (int i = 0; i < 8; ++i) koff[i] = (i >> 2) * 16 + half * 8 + (i & 3) * 2;

  // stage one 128x32 K-tile of A and Wt into buffer `buf` (async, 16B chunks)
  auto stage = [&](int buf, int kt) {
#pragma unroll
    for (int it = 0; it < 2; ++it) {
      int cid = tid + it * 256;              // 512 chunks of 8 bf16
      int row = cid >> 2, col = (cid & 3) * 8;
      async_b128(lds_off(&As[buf][row * 40 + col]), A,
                 (unsigned)(((m0 + row) * K + kt + col) * 2));
      async_b128(lds_off(&Bs[buf][row * 40 + col]), Wt,
                 (unsigned)(((n0 + row) * K + kt + col) * 2));
    }
  };

  const int nt = K / 32;
  stage(0, 0);
  for (int t = 0; t < nt; ++t) {
    wait_async0();            // my chunks of buffer t landed
    __syncthreads();          // everyone's chunks landed; prev compute done
    if (t + 1 < nt) stage((t + 1) & 1, (t + 1) * 32);   // overlaps WMMAs below

    const unsigned short* Ab = As[t & 1];
    const unsigned short* Bb = Bs[t & 1];
    FragAB a[2], b[4];
#pragma unroll
    for (int im = 0; im < 2; ++im) {
      int mrow = wm * 32 + im * 16 + lm;
#pragma unroll
      for (int i = 0; i < 8; ++i)
        a[im].u[i] = *(const unsigned*)&Ab[mrow * 40 + koff[i]];
    }
#pragma unroll
    for (int in = 0; in < 4; ++in) {
      int nrow = wn * 64 + in * 16 + lm;
#pragma unroll
      for (int i = 0; i < 8; ++i)
        b[in].u[i] = *(const unsigned*)&Bb[nrow * 40 + koff[i]];
    }
#pragma unroll
    for (int im = 0; im < 2; ++im)
#pragma unroll
      for (int in = 0; in < 4; ++in)
        acc[im][in] = wmma_bf16(a[im], b[in], acc[im][in]);
  }

  // epilogue: C frag element r -> M = r + 8*half, N = lm
#pragma unroll
  for (int im = 0; im < 2; ++im)
#pragma unroll
    for (int in = 0; in < 4; ++in)
#pragma unroll
      for (int r = 0; r < 8; ++r) {
        int row = m0 + wm * 32 + im * 16 + r + 8 * half;
        int col = n0 + wn * 64 + in * 16 + lm;
        float v = acc[im][in][r];
        if (MODE == 3) {
          ((float*)outp)[(size_t)row * N + col] = v;
        } else {
          unsigned short* o = (unsigned short*)outp;
          int bb = row >> 11, t = row & (T_ - 1);
          int hh = col >> 7,  hd = col & (HD - 1);
          if (MODE == 0)
            o[(((size_t)bb * NH + hh) * T_ + t) * HD + hd] = f2bf(v);
          else if (MODE == 1)
            o[(((size_t)bb * NKV + hh) * T_ + t) * HD + hd] = f2bf(v);
          else
            o[(((size_t)bb * NKV + hh) * HD + hd) * T_ + t] = f2bf(v);
        }
      }
}

// ---------------------------------------------------------------- RoPE (in place)
__global__ __launch_bounds__(256) void k_rope(unsigned short* buf, int nbh) {
  int idx = blockIdx.x * 256 + threadIdx.x;
  if (idx >= nbh * T_ * (HD / 2)) return;
  int d  = idx & 63;
  int t  = (idx >> 6) & (T_ - 1);
  int bh = idx >> 17;
  unsigned short* p = buf + ((size_t)bh * T_ + t) * HD;
  float inv = __powf(10000.f, -(float)d * (1.f / 64.f));
  float ang = (float)t * inv;
  float c = __cosf(ang), s = __sinf(ang);
  float x0 = bf2f(p[d]), x1 = bf2f(p[d + 64]);
  p[d]      = f2bf(x0 * c - x1 * s);
  p[d + 64] = f2bf(x1 * c + x0 * s);
}

// ---------------------------------------------------------------- flash attention
// block = (b, h, 128-query tile); 8 waves x 16 queries; 32-key inner tiles.
// K/V^T tiles double-buffered via async-to-LDS; staging overlaps S/PV WMMAs.
__global__ __launch_bounds__(256) void k_attn(const unsigned short* __restrict__ Q,
                                              const unsigned short* __restrict__ Kc,
                                              const unsigned short* __restrict__ Vt,
                                              unsigned short* __restrict__ AO) {
  __shared__ unsigned short Ks[2][32 * 136];   // [key][hd], stride 136 (272B rows)
  __shared__ unsigned short Vs[2][128 * 40];   // [hd][key], stride 40  (80B rows)
  __shared__ unsigned short Ps[8 * 16 * 36];   // per-wave P scratch [m][key]

  const int tid  = threadIdx.x;
  const int lane = tid & 31, w = tid >> 5;
  const int half = lane >> 4, lm = lane & 15;
  const int bh = blockIdx.x;
  const int b = bh >> 4, h = bh & 15, hk = h >> 2;
  const int q0 = blockIdx.y * 128;
  const float scale = 0.08838834764831845f;    // 1/sqrt(128)

  const unsigned short* qg = Q  + ((size_t)b * NH  + h ) * T_ * HD;
  const unsigned short* kg = Kc + ((size_t)b * NKV + hk) * T_ * HD;
  const unsigned short* vg = Vt + ((size_t)b * NKV + hk) * HD * T_;
  unsigned short* Pw = Ps + w * 16 * 36;

  // Q fragments (M = query, K = hd) live in registers for the whole loop
  FragAB qf[4];
#pragma unroll
  for (int kc = 0; kc < 4; ++kc)
#pragma unroll
    for (int i = 0; i < 8; ++i) {
      int ko = kc * 32 + (i >> 2) * 16 + half * 8 + (i & 3) * 2;
      qf[kc].u[i] = *(const unsigned*)&qg[(q0 + w * 16 + lm) * HD + ko];
    }

  v8f o[8];
#pragma unroll
  for (int n = 0; n < 8; ++n) o[n] = vzero8();
  float mrun[8], lrun[8];
#pragma unroll
  for (int r = 0; r < 8; ++r) { mrun[r] = -1e30f; lrun[r] = 0.f; }

  auto stageKV = [&](int buf, int kt) {
#pragma unroll
    for (int it = 0; it < 2; ++it) {
      int cid = tid + it * 256;
      int kr = cid >> 4, kc2 = (cid & 15) * 8;      // K tile: 32 x 128
      async_b128(lds_off(&Ks[buf][kr * 136 + kc2]), kg,
                 (unsigned)(((kt + kr) * HD + kc2) * 2));
      int vr = cid >> 2, vc = (cid & 3) * 8;        // V^T tile: 128 x 32
      async_b128(lds_off(&Vs[buf][vr * 40 + vc]), vg,
                 (unsigned)((vr * T_ + kt + vc) * 2));
    }
  };

  const int ntiles = (q0 + 128) / 32;
  stageKV(0, 0);
  for (int ti = 0; ti < ntiles; ++ti) {
    const int kt = ti * 32;
    wait_async0();
    __syncthreads();
    if (ti + 1 < ntiles) stageKV((ti + 1) & 1, kt + 32);

    const unsigned short* Kb = Ks[ti & 1];
    const unsigned short* Vb = Vs[ti & 1];

    // S = Q K^T  (two 16x16 frags, chained over hd)
    v8f s0 = vzero8(), s1 = vzero8();
#pragma unroll
    for (int kc = 0; kc < 4; ++kc) {
      FragAB b0, b1;
#pragma unroll
      for (int i = 0; i < 8; ++i) {
        int c = kc * 32 + (i >> 2) * 16 + half * 8 + (i & 3) * 2;
        b0.u[i] = *(const unsigned*)&Kb[lm * 136 + c];
        b1.u[i] = *(const unsigned*)&Kb[(16 + lm) * 136 + c];
      }
      s0 = wmma_bf16(qf[kc], b0, s0);
      s1 = wmma_bf16(qf[kc], b1, s1);
    }

    // online softmax (C-layout rows: M = r + 8*half, N = lane%16)
    float cfac[8];
#pragma unroll
    for (int r = 0; r < 8; ++r) {
      int qrow = q0 + w * 16 + r + 8 * half;
      int k0 = kt + lm, k1 = kt + 16 + lm;
      float a0 = (k0 <= qrow) ? s0[r] * scale : -1e30f;
      float a1 = (k1 <= qrow) ? s1[r] * scale : -1e30f;
      float tm = fmaxf(a0, a1);
#pragma unroll
      for (int off = 1; off < 16; off <<= 1) tm = fmaxf(tm, __shfl_xor(tm, off, 32));
      float mnew = fmaxf(mrun[r], tm);
      float corr = __expf(mrun[r] - mnew);
      float p0 = (k0 <= qrow) ? __expf(a0 - mnew) : 0.f;
      float p1 = (k1 <= qrow) ? __expf(a1 - mnew) : 0.f;
      float rs = p0 + p1;
#pragma unroll
      for (int off = 1; off < 16; off <<= 1) rs += __shfl_xor(rs, off, 32);
      lrun[r] = lrun[r] * corr + rs;
      mrun[r] = mnew;
      cfac[r] = corr;
      Pw[(r + 8 * half) * 36 + lm]      = f2bf(p0);
      Pw[(r + 8 * half) * 36 + 16 + lm] = f2bf(p1);
    }
#pragma unroll
    for (int n = 0; n < 8; ++n)
#pragma unroll
      for (int r = 0; r < 8; ++r) o[n][r] *= cfac[r];

    // wave-local LDS RAW on P scratch (DS ops in-order per wave)
    asm volatile("s_wait_dscnt 0" ::: "memory");

    FragAB pf;
#pragma unroll
    for (int i = 0; i < 8; ++i) {
      int ko = (i >> 2) * 16 + half * 8 + (i & 3) * 2;
      pf.u[i] = *(const unsigned*)&Pw[lm * 36 + ko];
    }
#pragma unroll
    for (int n = 0; n < 8; ++n) {
      FragAB vb;
#pragma unroll
      for (int i = 0; i < 8; ++i)
        vb.u[i] = *(const unsigned*)&Vb[(n * 16 + lm) * 40 + (i >> 2) * 16 + half * 8 + (i & 3) * 2];
      o[n] = wmma_bf16(pf, vb, o[n]);
    }
  }

  // epilogue: AO[b][t][h*128+hd] (bf16, feeds output projection)
#pragma unroll
  for (int n = 0; n < 8; ++n)
#pragma unroll
    for (int r = 0; r < 8; ++r) {
      int tq = q0 + w * 16 + r + 8 * half;
      int hd = n * 16 + lm;
      AO[((size_t)b * T_ + tq) * D_ + h * HD + hd] = f2bf(o[n][r] / lrun[r]);
    }
}

// ---------------------------------------------------------------- launch
extern "C" void kernel_launch(void* const* d_in, const int* in_sizes, int n_in,
                              void* d_out, int out_size, void* d_ws, size_t ws_size,
                              hipStream_t stream) {
  (void)in_sizes; (void)n_in; (void)out_size; (void)ws_size;
  const float* x  = (const float*)d_in[0];
  const float* wq = (const float*)d_in[1];
  const float* wk = (const float*)d_in[2];
  const float* wv = (const float*)d_in[3];
  const float* wo = (const float*)d_in[4];

  unsigned short* p = (unsigned short*)d_ws;
  const size_t nX  = (size_t)B_ * T_ * D_;        // 8388608
  const size_t nWq = (size_t)D_ * NH * HD;        // 4194304
  const size_t nWk = (size_t)D_ * NKV * HD;       // 1048576
  const size_t nKV = (size_t)B_ * NKV * T_ * HD;  // 2097152
  unsigned short* xb  = p; p += nX;
  unsigned short* wqt = p; p += nWq;   // [N][K] transposed bf16 weights
  unsigned short* wkt = p; p += nWk;
  unsigned short* wvt = p; p += nWk;
  unsigned short* wot = p; p += nWq;
  unsigned short* Qb  = p; p += nX;
  unsigned short* Kb  = p; p += nKV;
  unsigned short* Vtb = p; p += nKV;
  unsigned short* AOb = p; p += nX;

  k_f32_to_bf16<<<(unsigned)(nX / 1024), 256, 0, stream>>>(x, xb, (int)nX);
  k_f32_to_bf16_T<<<(unsigned)(nWq / 256), 256, 0, stream>>>(wq, wqt, NH * HD,  D_);
  k_f32_to_bf16_T<<<(unsigned)(nWk / 256), 256, 0, stream>>>(wk, wkt, NKV * HD, D_);
  k_f32_to_bf16_T<<<(unsigned)(nWk / 256), 256, 0, stream>>>(wv, wvt, NKV * HD, D_);
  k_f32_to_bf16_T<<<(unsigned)(nWq / 256), 256, 0, stream>>>(wo, wot, D_,       D_);

  dim3 blk(256);
  k_gemm<0><<<dim3(16, 32), blk, 0, stream>>>(xb, wqt, Qb,  NH * HD,  D_);
  k_gemm<1><<<dim3(4,  32), blk, 0, stream>>>(xb, wkt, Kb,  NKV * HD, D_);
  k_gemm<2><<<dim3(4,  32), blk, 0, stream>>>(xb, wvt, Vtb, NKV * HD, D_);

  k_rope<<<(B_ * NH  * T_ * (HD / 2)) / 256, 256, 0, stream>>>(Qb, B_ * NH);
  k_rope<<<(B_ * NKV * T_ * (HD / 2)) / 256, 256, 0, stream>>>(Kb, B_ * NKV);

  k_attn<<<dim3(B_ * NH, T_ / 128), blk, 0, stream>>>(Qb, Kb, Vtb, AOb);

  k_gemm<3><<<dim3(16, 32), blk, 0, stream>>>(AOb, wot, d_out, D_, D_);
}
